// Qwen3MoeSparseMoeBlock_37838661878033
// MI455X (gfx1250) — matrix operand order repaired and checked by reading, not compiled, not run on hardware
//
#include <hip/hip_runtime.h>

// ---------------------------------------------------------------------------
// Qwen3 MoE sparse block for gfx1250 (MI455X).
// bf16 WMMA (f32 accumulate) + TDM tensor_load_to_lds staging + prefetch.
// Sizes fixed by the reference: B=2,S=1024 -> T=2048 tokens, D=2048, E=8, F=768
// ---------------------------------------------------------------------------

typedef __attribute__((ext_vector_type(16))) __bf16   v16bf;
typedef __attribute__((ext_vector_type(8)))  float    v8f;
typedef __attribute__((ext_vector_type(4)))  unsigned u32x4;
typedef __attribute__((ext_vector_type(8)))  int      i32x8;
typedef __attribute__((ext_vector_type(4)))  int      i32x4;

#define T_TOK 2048
#define DIM   2048
#define NEXP  8
#define FFN   768

// LDS byte offset of a __shared__ object (AS(3) pointers are 32-bit offsets)
#define LDS_OFF(p) ((unsigned)(size_t)(__attribute__((address_space(3))) const void*)(p))

// ---------------------------------------------------------------------------
// TDM: 2-D tile load Global -> LDS via tensor descriptor (ISA ch.8 D# layout).
//   row_len : tensor dim0 length (elements)  -> also dim0 stride (dense rows)
//   n_rows  : tensor dim1 length (rows)
//   tile_c  : tile dim0 (elements per row of the tile)
//   tile_r  : tile dim1 (rows of the tile)
//   dsz     : data_size code (0=1B,1=2B,2=4B,3=8B)
// Issue from ONE wave only (TENSORcnt is per-wave; EXEC ignored by tensor ops)
// ---------------------------------------------------------------------------
__device__ __forceinline__ void tdm_load_2d(unsigned lds_off, const void* gptr,
                                            unsigned row_len, unsigned n_rows,
                                            unsigned tile_c, unsigned tile_r,
                                            unsigned dsz)
{
    const unsigned long long ga = (unsigned long long)(size_t)gptr;

    u32x4 g0;
    g0[0] = 1u;                                           // count=1, user mode
    g0[1] = lds_off;                                      // lds_addr
    g0[2] = (unsigned)ga;                                 // global_addr[31:0]
    g0[3] = (unsigned)((ga >> 32) & 0x1FFFFFFull)         // global_addr[56:32]
          | (2u << 30);                                   // type=2 ("image")

    i32x8 g1;
    g1[0] = (int)(dsz << 16);                             // wg_mask=0, data_size
    g1[1] = (int)((row_len & 0xFFFFu) << 16);             // tensor_dim0[15:0]
    g1[2] = (int)(((row_len >> 16) & 0xFFFFu)             // tensor_dim0[31:16]
          |        ((n_rows & 0xFFFFu) << 16));           // tensor_dim1[15:0]
    g1[3] = (int)(((n_rows >> 16) & 0xFFFFu)              // tensor_dim1[31:16]
          |        (tile_c << 16));                       // tile_dim0
    g1[4] = (int)tile_r;                                  // tile_dim1 (dim2=0)
    g1[5] = (int)row_len;                                 // dim0_stride[31:0]
    g1[6] = 0;                                            // stride hi / dim1_stride
    g1[7] = 0;

    const i32x4 z4 = {};
    const i32x8 z8 = {};
    // 6-arg form on this toolchain: (g0, g1, g2, g3, g4, cpol); 2-D tensor ->
    // trailing groups zero-filled.
    __builtin_amdgcn_tensor_load_to_lds(g0, g1, z4, z4, z8, 0);
}

// ---------------------------------------------------------------------------
// Kernel 1: router logits + softmax + top-2 + renormalized combine weights.
// One wave32 per token.
// ---------------------------------------------------------------------------
__global__ void moe_router(const float* __restrict__ x,
                           const float* __restrict__ gate_w,
                           float* __restrict__ logits,     // [T, E]
                           float* __restrict__ combine)    // [T, E] dense weights
{
    const int wave = threadIdx.x >> 5;
    const int lane = threadIdx.x & 31;
    const int t = blockIdx.x * 8 + wave;
    if (t >= T_TOK) return;

    float acc[NEXP];
    #pragma unroll
    for (int e = 0; e < NEXP; ++e) acc[e] = 0.f;

    const float* xr = x + (size_t)t * DIM;
    for (int d = lane; d < DIM; d += 32) {
        const float xv = xr[d];
        const float* gw = gate_w + (size_t)d * NEXP;   // row-major [D, E]
        #pragma unroll
        for (int e = 0; e < NEXP; ++e) acc[e] += xv * gw[e];
    }
    #pragma unroll
    for (int off = 16; off > 0; off >>= 1) {
        #pragma unroll
        for (int e = 0; e < NEXP; ++e)
            acc[e] += __shfl_xor(acc[e], off, 32);
    }

    if (lane == 0) {
        float m = acc[0];
        #pragma unroll
        for (int e = 1; e < NEXP; ++e) m = fmaxf(m, acc[e]);
        float p[NEXP]; float s = 0.f;
        #pragma unroll
        for (int e = 0; e < NEXP; ++e) { p[e] = __expf(acc[e] - m); s += p[e]; }
        int i1 = 0;                                   // ties -> lowest index
        #pragma unroll
        for (int e = 1; e < NEXP; ++e) if (p[e] > p[i1]) i1 = e;
        int i2 = (i1 == 0) ? 1 : 0;
        #pragma unroll
        for (int e = 0; e < NEXP; ++e) if (e != i1 && p[e] > p[i2]) i2 = e;
        float w1 = p[i1] / s, w2 = p[i2] / s;
        const float tot = w1 + w2;                    // norm_topk_prob = True
        w1 /= tot; w2 /= tot;
        #pragma unroll
        for (int e = 0; e < NEXP; ++e) {
            logits [(size_t)t * NEXP + e] = acc[e];
            combine[(size_t)t * NEXP + e] = (e == i1) ? w1 : (e == i2) ? w2 : 0.f;
        }
    }
}

// ---------------------------------------------------------------------------
// A-fragment builders (ISA 7.12.2, 16-bit A 16x32):
// lane l: m = l&15, khi = l>>4; elems 0..7 -> K = khi*8+i, 8..15 -> 16+khi*8+i
// ---------------------------------------------------------------------------
__device__ __forceinline__ v16bf a_frag_from_f32(const float* __restrict__ Asf,
                                                 int lane)
{
    const int ln = lane & 15, khi = lane >> 4;
    const float* rp = Asf + ln * 32;
    v16bf a;
    #pragma unroll
    for (int i = 0; i < 8; ++i)  a[i]     = (__bf16)rp[khi * 8 + i];
    #pragma unroll
    for (int i = 0; i < 8; ++i)  a[8 + i] = (__bf16)rp[16 + khi * 8 + i];
    return a;
}

__device__ __forceinline__ v16bf a_frag_from_bf16(const __bf16* __restrict__ As,
                                                  int lane)
{
    const int ln = lane & 15, khi = lane >> 4;
    const __bf16* rp = As + ln * 32;
    v16bf a;
    #pragma unroll
    for (int i = 0; i < 8; ++i)  a[i]     = rp[khi * 8 + i];
    #pragma unroll
    for (int i = 0; i < 8; ++i)  a[8 + i] = rp[16 + khi * 8 + i];
    return a;
}

// ---------------------------------------------------------------------------
// Kernel 2 (per expert): g = x@Wg, u = x@Wu (bf16 WMMA), h = silu(g)*u -> bf16.
// Block = 256 threads = 8 waves covering 16(T) x 128(F); one 16x16 tile/wave.
// x K-chunk (16x32 f32) staged to LDS by the Tensor Data Mover.
// ---------------------------------------------------------------------------
__global__ void moe_gateup(const float* __restrict__ x,
                           const float* __restrict__ wg,    // [D, F] of expert e
                           const float* __restrict__ wu,    // [D, F] of expert e
                           __bf16* __restrict__ hbuf)       // [T, F]
{
    __shared__ __align__(16) float Asf[16 * 32];            // raw f32 x tile

    const int wave = threadIdx.x >> 5;
    const int lane = threadIdx.x & 31;
    const int ln   = lane & 15;
    const int khi  = lane >> 4;

    const int FG   = FFN / 128;                 // 6 f-groups of 8 tiles
    const int fgrp = blockIdx.x % FG;
    const int mt   = blockIdx.x / FG;           // 0..127
    const int t0   = mt * 16;
    const int n0   = (fgrp * 8 + wave) * 16;

    const unsigned lds_a = LDS_OFF(Asf);

    v8f accg = {};
    v8f accu = {};

    for (int k0 = 0; k0 < DIM; k0 += 32) {
        __syncthreads();                        // tile consumed by all waves
        if (wave == 0) {
            tdm_load_2d(lds_a, x + (size_t)t0 * DIM + k0,
                        DIM, T_TOK, /*tile_c=*/32, /*tile_r=*/16, /*dsz f32*/2);
            __builtin_amdgcn_s_wait_tensorcnt(0);
        }
        // prefetch next K-chunk's weight rows (one row per lane)
        if (k0 + 32 < DIM) {
            const size_t poff = (size_t)(k0 + 32 + lane) * FFN + n0;
            __builtin_prefetch(wg + poff, 0, 1);
            __builtin_prefetch(wu + poff, 0, 1);
        }
        __syncthreads();                        // publish TDM tile to all waves

        const v16bf a = a_frag_from_f32(Asf, lane);

        // B frags: elems 0..7 -> K = khi*8+i, 8..15 -> 16+khi*8+(i-8), N = lane&15
        v16bf bg, bu;
        #pragma unroll
        for (int i = 0; i < 16; ++i) {
            const int kk = k0 + (i < 8 ? khi * 8 + i : 16 + khi * 8 + (i - 8));
            const size_t off = (size_t)kk * FFN + n0 + ln;
            bg[i] = (__bf16)wg[off];
            bu[i] = (__bf16)wu[off];
        }

        accg = __builtin_amdgcn_wmma_f32_16x16x32_bf16(
                   false, a, false, bg, (short)0, accg, false, false);
        accu = __builtin_amdgcn_wmma_f32_16x16x32_bf16(
                   false, a, false, bu, (short)0, accu, false, false);
    }

    // h = silu(g) * u ; C/D layout: elem j -> M = khi*8 + j, N = lane&15
    #pragma unroll
    for (int j = 0; j < 8; ++j) {
        const int m = khi * 8 + j;
        const float g = accg[j];
        const float u = accu[j];
        const float h = (g / (1.f + __expf(-g))) * u;
        hbuf[(size_t)(t0 + m) * FFN + n0 + ln] = (__bf16)h;
    }
}

// ---------------------------------------------------------------------------
// Kernel 3 (per expert): y = h @ Wd, rows scaled by combine[:,e], write/acc out.
// Block = 256 threads = 8 waves covering 16(T) x 128(D); h chunk via TDM.
// ---------------------------------------------------------------------------
__global__ void moe_down(const __bf16* __restrict__ hbuf,   // [T, F] bf16
                         const float* __restrict__ wd,      // [F, D] of expert e
                         const float* __restrict__ combine, // [T, E]
                         float* __restrict__ out,           // [T, D]
                         int e, int first)
{
    __shared__ __align__(16) __bf16 As[16 * 32];
    __shared__ float Cw[16];

    const int wave = threadIdx.x >> 5;
    const int lane = threadIdx.x & 31;
    const int ln   = lane & 15;
    const int khi  = lane >> 4;

    const int DG   = DIM / 128;                 // 16 d-groups of 8 tiles
    const int dgrp = blockIdx.x % DG;
    const int mt   = blockIdx.x / DG;           // 0..127
    const int t0   = mt * 16;
    const int n0   = (dgrp * 8 + wave) * 16;

    if (threadIdx.x < 16)
        Cw[threadIdx.x] = combine[(size_t)(t0 + threadIdx.x) * NEXP + e];

    const unsigned lds_a = LDS_OFF(As);

    v8f ce = {};
    for (int f0 = 0; f0 < FFN; f0 += 32) {
        __syncthreads();
        if (wave == 0) {
            tdm_load_2d(lds_a, hbuf + (size_t)t0 * FFN + f0,
                        FFN, T_TOK, /*tile_c=*/32, /*tile_r=*/16, /*dsz bf16*/1);
            __builtin_amdgcn_s_wait_tensorcnt(0);
        }
        if (f0 + 32 < FFN) {
            __builtin_prefetch(wd + (size_t)(f0 + 32 + lane) * DIM + n0, 0, 1);
        }
        __syncthreads();

        const v16bf a = a_frag_from_bf16(As, lane);

        v16bf b;
        #pragma unroll
        for (int i = 0; i < 16; ++i) {
            const int kk = f0 + (i < 8 ? khi * 8 + i : 16 + khi * 8 + (i - 8));
            b[i] = (__bf16)wd[(size_t)kk * DIM + n0 + ln];
        }
        ce = __builtin_amdgcn_wmma_f32_16x16x32_bf16(
                 false, a, false, b, (short)0, ce, false, false);
    }

    #pragma unroll
    for (int j = 0; j < 8; ++j) {
        const int m    = khi * 8 + j;
        const size_t o = (size_t)(t0 + m) * DIM + n0 + ln;
        const float v  = Cw[m] * ce[j];
        out[o] = first ? v : (out[o] + v);
    }
}

// ---------------------------------------------------------------------------
// Launch: router once, then per-expert gateup+down (stream-ordered; the single
// [T,F] bf16 h-buffer in d_ws is reused and stays resident in the 192MB L2).
// ---------------------------------------------------------------------------
extern "C" void kernel_launch(void* const* d_in, const int* in_sizes, int n_in,
                              void* d_out, int out_size, void* d_ws, size_t ws_size,
                              hipStream_t stream)
{
    (void)in_sizes; (void)n_in; (void)out_size; (void)ws_size;

    const float* x      = (const float*)d_in[0];   // [B,S,D] -> [T,D]
    const float* gate_w = (const float*)d_in[1];   // [D,E]
    const float* w_gate = (const float*)d_in[2];   // [E,D,F]
    const float* w_up   = (const float*)d_in[3];   // [E,D,F]
    const float* w_down = (const float*)d_in[4];   // [E,F,D]

    float* out    = (float*)d_out;                       // [T,D]
    float* logits = out + (size_t)T_TOK * DIM;           // [T,E] (2nd output)

    float*  combine = (float*)d_ws;                                    // T*E f32
    __bf16* hbuf    = (__bf16*)((char*)d_ws +
                                (size_t)T_TOK * NEXP * sizeof(float)); // T*F bf16

    moe_router<<<T_TOK / 8, 256, 0, stream>>>(x, gate_w, logits, combine);

    const int gu_grid   = (T_TOK / 16) * (FFN / 128);   // 128 * 6  = 768
    const int down_grid = (T_TOK / 16) * (DIM / 128);   // 128 * 16 = 2048

    for (int e = 0; e < NEXP; ++e) {
        const float* wg = w_gate + (size_t)e * DIM * FFN;
        const float* wu = w_up   + (size_t)e * DIM * FFN;
        const float* wd = w_down + (size_t)e * FFN * DIM;
        moe_gateup<<<gu_grid, 256, 0, stream>>>(x, wg, wu, hbuf);
        moe_down  <<<down_grid, 256, 0, stream>>>(hbuf, wd, combine, out,
                                                  e, e == 0 ? 1 : 0);
    }
}
